// MixtureOfBlockAttention_2052994367913
// MI455X (gfx1250) — matrix operand order repaired and checked by reading, not compiled, not run on hardware
//
#include <hip/hip_runtime.h>
#include <hip/hip_bf16.h>
#include <stdint.h>

// ---------------- problem constants ----------------
#define S      4096
#define DIM    2048
#define NH     16
#define NKV    8
#define HD     128
#define NBLK   32            // S / 128
#define KVW    (NKV*HD)      // 1024
#define QW     (NH*HD)       // 2048
#define NEG_M  (-10000.0f)
#define INV_SQRT_D 0.08838834764831845f

// ---------------- WMMA fragment types (CDNA5, wave32) ----------------
typedef __attribute__((ext_vector_type(16))) __bf16    v16bf;
typedef __attribute__((ext_vector_type(8)))  float     v8f;
typedef __attribute__((ext_vector_type(4)))  uint32_t  v4u;

union AFragU { v16bf v; v4u q[2]; };

__device__ __forceinline__ float bf2f(uint16_t h) {
    uint32_t u = ((uint32_t)h) << 16; float f;
    __builtin_memcpy(&f, &u, 4); return f;
}
__device__ __forceinline__ uint16_t f2bf(float f) {
    uint32_t u; __builtin_memcpy(&u, &f, 4);
    uint32_t r = u + 0x7FFFu + ((u >> 16) & 1u);   // round-to-nearest-even
    return (uint16_t)(r >> 16);
}

// A-matrix 16x32 bf16 fragment (M x K), row stride ld (elements).
// ISA layout: lane L: m=L%16, half=L/16; VGPR0-3 hold K = half*8 .. half*8+7,
// VGPR4-7 hold K = 16+half*8 .. 16+half*8+7  -> two contiguous 16B loads.
__device__ __forceinline__ v16bf load_a16x32(const uint16_t* base, int ld) {
    int lane = threadIdx.x & 31;
    int m = lane & 15, half = lane >> 4;
    const uint16_t* p = base + m * ld + half * 8;
    AFragU a;
    a.q[0] = *(const v4u*)(p);
    a.q[1] = *(const v4u*)(p + 16);
    return a.v;
}

// B-matrix 32x16 bf16 fragment (K x N) given B-transposed storage:
// baseT points at BT row 0 (column 0 of B), row stride ld.
// ISA layout: lane L: n=L%16, half=L/16; VGPR0-3 hold K = half*8 .. half*8+7,
// VGPR4-7 hold K = 16+half*8 .. 16+half*8+7  -> two contiguous 16B loads.
__device__ __forceinline__ v16bf load_b32x16(const uint16_t* baseT, int ld) {
    int lane = threadIdx.x & 31;
    int n = lane & 15, half = lane >> 4;
    const uint16_t* p = baseT + n * ld + half * 8;
    AFragU b;
    b.q[0] = *(const v4u*)(p);
    b.q[1] = *(const v4u*)(p + 16);
    return b.v;
}

__device__ __forceinline__ v8f wmma_bf16(v16bf a, v16bf b, v8f c) {
    return __builtin_amdgcn_wmma_f32_16x16x32_bf16(false, a, false, b, (short)0, c, false, false);
}

// ---------------- conversion / layout kernels ----------------
__global__ __launch_bounds__(256, 1)
void cvt_f32_to_bf16(const float* __restrict__ in, uint16_t* __restrict__ out, int n) {
    int i = blockIdx.x * blockDim.x + threadIdx.x;
    if (i < n) out[i] = f2bf(in[i]);
}

// wT[n*K + k] = bf16(w[k*N + n])
__global__ __launch_bounds__(256, 1)
void transpose_w(const float* __restrict__ w, uint16_t* __restrict__ wT, int K, int N) {
    int i = blockIdx.x * blockDim.x + threadIdx.x;
    if (i >= K * N) return;
    int n = i / K, k = i - n * K;
    wT[i] = f2bf(w[(size_t)k * N + n]);
}

// in-place interleaved RoPE on bf16 [S][nheads*128]
__global__ __launch_bounds__(256, 1)
void rope_kernel(uint16_t* __restrict__ t, int nheads) {
    int i = blockIdx.x * blockDim.x + threadIdx.x;
    int total = S * nheads * 64;
    if (i >= total) return;
    int pi = i & 63;
    int rest = i >> 6;
    int hh = rest % nheads;
    int s  = rest / nheads;
    int ld = nheads * HD;
    uint16_t* p = t + (size_t)s * ld + hh * HD + 2 * pi;
    float t1 = bf2f(p[0]), t2 = bf2f(p[1]);
    float inv = __powf(10000.0f, -((float)(2 * pi)) / (float)HD);
    float ang = (float)s * inv;
    float sn, cs; __sincosf(ang, &sn, &cs);
    p[0] = f2bf(t1 * cs - t2 * sn);
    p[1] = f2bf(t1 * sn + t2 * cs);
}

// vT[kvh][d][s] = v[s][kvh*128+d]
__global__ __launch_bounds__(256, 1)
void vtrans_kernel(const uint16_t* __restrict__ vb, uint16_t* __restrict__ vT) {
    int i = blockIdx.x * blockDim.x + threadIdx.x;
    if (i >= S * KVW) return;
    int s = i >> 10;
    int c = i & 1023;
    vT[((size_t)c << 12) + s] = vb[i];   // c = kvh*128+d ; <<12 = *S
}

// bm[kvh*32+blk][d] = mean over 128 tokens of post-RoPE K
__global__ __launch_bounds__(128, 1)
void blockmean_kernel(const uint16_t* __restrict__ kb, float* __restrict__ bm) {
    int kvh = blockIdx.x >> 5, blk = blockIdx.x & 31, d = threadIdx.x;
    const uint16_t* base = kb + (size_t)(blk * 128) * KVW + kvh * HD + d;
    float acc = 0.f;
#pragma unroll 8
    for (int t = 0; t < 128; ++t) acc += bf2f(base[(size_t)t * KVW]);
    bm[(kvh * NBLK + blk) * HD + d] = acc * (1.0f / 128.0f);
}

// one wave per (head, token): gating dot + future mask + top-8 -> 32-bit block mask
__global__ __launch_bounds__(256, 1)
void gating_topk_kernel(const uint16_t* __restrict__ qb, const float* __restrict__ bm,
                        uint32_t* __restrict__ selmask) {
    int wave = (blockIdx.x * blockDim.x + threadIdx.x) >> 5;
    int lane = threadIdx.x & 31;
    int h = wave >> 12;          // 4096 tokens per head
    int s = wave & (S - 1);
    if (h >= NH) return;
    int kvh = h >> 1;
    const uint16_t* q = qb + (size_t)s * QW + h * HD;
    const float*    b = bm + (size_t)(kvh * NBLK + lane) * HD;
    float acc = 0.f;
#pragma unroll 8
    for (int d = 0; d < HD; ++d) acc = fmaf(bf2f(q[d]), b[d], acc);
    float g = acc * INV_SQRT_D;
    if (lane > (s >> 7)) g = NEG_M;             // future-block gating mask
    uint32_t bits = 0;
#pragma unroll
    for (int it = 0; it < 8; ++it) {            // TOPK = 8, lowest-index tie break
        float bv = g; int bi = lane;
#pragma unroll
        for (int m = 16; m; m >>= 1) {
            float ov = __shfl_xor(bv, m, 32);
            int   oi = __shfl_xor(bi, m, 32);
            if (ov > bv || (ov == bv && oi < bi)) { bv = ov; bi = oi; }
        }
        bits |= (1u << bi);
        if (lane == bi) g = -3.0e38f;
    }
    if (lane == 0) selmask[(size_t)h * S + s] = bits;
}

// ---------------- bf16 WMMA GEMM: C[MxN] = A[MxK] * B, B given transposed ----------------
// 32x64 tile per wave (2x4 accumulators = 64 VGPRs),
// 8 v_wmma per 32-wide K step with 4x A-frag reuse and 2x B-frag reuse.
template <bool OUTF32>
__global__ __launch_bounds__(128, 1)
void gemm_bf16_kernel(const uint16_t* __restrict__ A, const uint16_t* __restrict__ BT,
                      void* __restrict__ C, int M, int N, int K) {
    int wave = (blockIdx.x * blockDim.x + threadIdx.x) >> 5;
    int lane = threadIdx.x & 31;
    int tilesN = N >> 6;
    int tm = (wave / tilesN) << 5;   // 32 rows
    int tn = (wave % tilesN) << 6;   // 64 cols
    if (tm >= M) return;

    v8f acc[2][4];
#pragma unroll
    for (int i = 0; i < 2; ++i)
#pragma unroll
        for (int j = 0; j < 4; ++j) { v8f z = {}; acc[i][j] = z; }

    for (int k0 = 0; k0 < K; k0 += 32) {
        v16bf a0 = load_a16x32(A + (size_t)tm * K + k0, K);
        v16bf a1 = load_a16x32(A + (size_t)(tm + 16) * K + k0, K);
        v16bf bfr[4];
#pragma unroll
        for (int j = 0; j < 4; ++j)
            bfr[j] = load_b32x16(BT + (size_t)(tn + 16 * j) * K + k0, K);
#pragma unroll
        for (int j = 0; j < 4; ++j) {
            acc[0][j] = wmma_bf16(a0, bfr[j], acc[0][j]);
            acc[1][j] = wmma_bf16(a1, bfr[j], acc[1][j]);
        }
    }

    int n = lane & 15, rbase = (lane >> 4) * 8;
#pragma unroll
    for (int i = 0; i < 2; ++i)
#pragma unroll
        for (int j = 0; j < 4; ++j)
#pragma unroll
            for (int r = 0; r < 8; ++r) {
                size_t idx = (size_t)(tm + 16 * i + rbase + r) * N + (tn + 16 * j + n);
                if (OUTF32) ((float*)C)[idx] = acc[i][j][r];
                else        ((uint16_t*)C)[idx] = f2bf(acc[i][j][r]);
            }
}

// ---------------- flash attention with MoBA mask ----------------
// one wave per (head, 16-row q tile); loops all 4096 keys in 32-wide chunks.
// mask rule (exactly reference): masked iff (t > s) AND block(t) not selected.
__global__ __launch_bounds__(128, 1)
void flash_kernel(const uint16_t* __restrict__ qb, const uint16_t* __restrict__ kb,
                  const uint16_t* __restrict__ vT, const uint32_t* __restrict__ selmask,
                  uint16_t* __restrict__ attn) {
    int wave  = (blockIdx.x * blockDim.x + threadIdx.x) >> 5;
    int lane  = threadIdx.x & 31;
    int wslot = (threadIdx.x >> 5) & 3;
    int h  = wave >> 8;            // 256 q tiles per head
    int qt = wave & 255;
    if (h >= NH) return;
    int q0  = qt << 4;
    int kvh = h >> 1;

    const uint16_t* qbase = qb + (size_t)q0 * QW + h * HD;
    const uint16_t* kbase = kb + kvh * HD;
    const uint16_t* vbase = vT + (size_t)kvh * HD * S;

    // resident Q fragments: 16x128 as 4 K-steps
    v16bf qa[4];
#pragma unroll
    for (int ks = 0; ks < 4; ++ks) qa[ks] = load_a16x32(qbase + ks * 32, QW);

    int rbase = (lane >> 4) * 8;   // D-fragment rows owned by this half-wave
    int ncol  = lane & 15;

    float mrow[8], lrow[8];
    uint32_t msk[8];
#pragma unroll
    for (int j = 0; j < 8; ++j) {
        int s_abs = q0 + rbase + j;
        msk[j]  = selmask[(size_t)h * S + s_abs];
        mrow[j] = -1.0e30f;
        lrow[j] = 0.0f;
    }
    v8f o[8];
#pragma unroll
    for (int d = 0; d < 8; ++d) { v8f z = {}; o[d] = z; }

    __shared__ alignas(16) uint16_t pb[4][16 * 32];
    uint16_t* pl = &pb[wslot][0];

    for (int t0 = 0; t0 < S; t0 += 32) {
        // ---- scores: two 16x16 tiles over d=128 (4 K-steps each) ----
        v8f s0 = {}, s1 = {};
#pragma unroll
        for (int ks = 0; ks < 4; ++ks) {
            v16bf b0 = load_b32x16(kbase + (size_t)t0 * KVW + ks * 32, KVW);
            v16bf b1 = load_b32x16(kbase + (size_t)(t0 + 16) * KVW + ks * 32, KVW);
            s0 = wmma_bf16(qa[ks], b0, s0);
            s1 = wmma_bf16(qa[ks], b1, s1);
        }
        // ---- mask + online softmax ----
        float p0[8], p1[8], scl[8];
#pragma unroll
        for (int j = 0; j < 8; ++j) {
            int s_abs = q0 + rbase + j;
            int ta = t0 + ncol, tb = t0 + 16 + ncol;
            float v0 = s0[j] * INV_SQRT_D;
            if (ta > s_abs && !((msk[j] >> (ta >> 7)) & 1u)) v0 += NEG_M;
            float v1 = s1[j] * INV_SQRT_D;
            if (tb > s_abs && !((msk[j] >> (tb >> 7)) & 1u)) v1 += NEG_M;
            float lm = fmaxf(v0, v1);
            lm = fmaxf(lm, __shfl_xor(lm, 1, 32));
            lm = fmaxf(lm, __shfl_xor(lm, 2, 32));
            lm = fmaxf(lm, __shfl_xor(lm, 4, 32));
            lm = fmaxf(lm, __shfl_xor(lm, 8, 32));
            float nm = fmaxf(mrow[j], lm);
            float sc = __expf(mrow[j] - nm);
            mrow[j] = nm; scl[j] = sc;
            float e0 = __expf(v0 - nm), e1 = __expf(v1 - nm);
            p0[j] = e0; p1[j] = e1;
            float ls = e0 + e1;
            ls += __shfl_xor(ls, 1, 32);
            ls += __shfl_xor(ls, 2, 32);
            ls += __shfl_xor(ls, 4, 32);
            ls += __shfl_xor(ls, 8, 32);
            lrow[j] = lrow[j] * sc + ls;
        }
#pragma unroll
        for (int d = 0; d < 8; ++d)
#pragma unroll
            for (int j = 0; j < 8; ++j) o[d][j] *= scl[j];

        // ---- P (D-layout) -> LDS -> A-layout fragment ----
#pragma unroll
        for (int j = 0; j < 8; ++j) {
            int r = rbase + j;
            pl[r * 32 + ncol]      = f2bf(p0[j]);
            pl[r * 32 + 16 + ncol] = f2bf(p1[j]);
        }
        v16bf pa = load_a16x32(pl, 32);   // same-wave DS ops are in-order

        // ---- P x V^T: 8 d-tiles ----
#pragma unroll
        for (int d = 0; d < 8; ++d) {
            v16bf bv = load_b32x16(vbase + (size_t)(d * 16) * S + t0, S);
            o[d] = wmma_bf16(pa, bv, o[d]);
        }
    }

    // epilogue: normalize, store bf16 into attn [S][NH*HD]
#pragma unroll
    for (int d = 0; d < 8; ++d) {
#pragma unroll
        for (int j = 0; j < 8; ++j) {
            float val = o[d][j] / lrow[j];
            int row = q0 + rbase + j;
            attn[(size_t)row * QW + h * HD + d * 16 + ncol] = f2bf(val);
        }
    }
}

// ---------------- host-side launcher ----------------
extern "C" void kernel_launch(void* const* d_in, const int* in_sizes, int n_in,
                              void* d_out, int out_size, void* d_ws, size_t ws_size,
                              hipStream_t stream) {
    const float* x  = (const float*)d_in[0];
    const float* wq = (const float*)d_in[1];
    const float* wk = (const float*)d_in[2];
    const float* wv = (const float*)d_in[3];
    const float* wo = (const float*)d_in[4];
    float* out = (float*)d_out;

    char* ws = (char*)d_ws;
    size_t off = 0;
    auto alloc = [&](size_t bytes) { char* p = ws + off; off += (bytes + 255) & ~(size_t)255; return p; };

    uint16_t* xb   = (uint16_t*)alloc((size_t)S * DIM * 2);
    uint16_t* wqT  = (uint16_t*)alloc((size_t)QW * DIM * 2);
    uint16_t* wkT  = (uint16_t*)alloc((size_t)KVW * DIM * 2);
    uint16_t* wvT  = (uint16_t*)alloc((size_t)KVW * DIM * 2);
    uint16_t* woT  = (uint16_t*)alloc((size_t)DIM * DIM * 2);
    uint16_t* qbuf = (uint16_t*)alloc((size_t)S * QW * 2);
    uint16_t* kbuf = (uint16_t*)alloc((size_t)S * KVW * 2);
    uint16_t* vbuf = (uint16_t*)alloc((size_t)S * KVW * 2);
    uint16_t* vT   = (uint16_t*)alloc((size_t)S * KVW * 2);
    float*    bm   = (float*)   alloc((size_t)NKV * NBLK * HD * 4);
    uint32_t* sel  = (uint32_t*)alloc((size_t)NH * S * 4);
    uint16_t* attn = (uint16_t*)alloc((size_t)S * QW * 2);

    // 1) convert activations + transpose/convert weights to bf16
    {
        int n = S * DIM;
        cvt_f32_to_bf16<<<(n + 255) / 256, 256, 0, stream>>>(x, xb, n);
        transpose_w<<<(DIM * QW  + 255) / 256, 256, 0, stream>>>(wq, wqT, DIM, QW);
        transpose_w<<<(DIM * KVW + 255) / 256, 256, 0, stream>>>(wk, wkT, DIM, KVW);
        transpose_w<<<(DIM * KVW + 255) / 256, 256, 0, stream>>>(wv, wvT, DIM, KVW);
        transpose_w<<<(DIM * DIM + 255) / 256, 256, 0, stream>>>(wo, woT, DIM, DIM);
    }
    // 2) QKV projections (WMMA): 32x64 tile per wave
    {
        int wavesQ = (S / 32) * (QW / 64);      // 4096
        gemm_bf16_kernel<false><<<wavesQ * 32 / 128, 128, 0, stream>>>(xb, wqT, qbuf, S, QW, DIM);
        int wavesK = (S / 32) * (KVW / 64);     // 2048
        gemm_bf16_kernel<false><<<wavesK * 32 / 128, 128, 0, stream>>>(xb, wkT, kbuf, S, KVW, DIM);
        gemm_bf16_kernel<false><<<wavesK * 32 / 128, 128, 0, stream>>>(xb, wvT, vbuf, S, KVW, DIM);
    }
    // 3) RoPE (q, k) and V transpose
    {
        int nq = S * NH * 64;
        rope_kernel<<<(nq + 255) / 256, 256, 0, stream>>>(qbuf, NH);
        int nk = S * NKV * 64;
        rope_kernel<<<(nk + 255) / 256, 256, 0, stream>>>(kbuf, NKV);
        int nv = S * KVW;
        vtrans_kernel<<<(nv + 255) / 256, 256, 0, stream>>>(vbuf, vT);
    }
    // 4) block means + gating top-8 masks
    {
        blockmean_kernel<<<NKV * NBLK, HD, 0, stream>>>(kbuf, bm);
        int waves = NH * S;                     // 65536 waves
        gating_topk_kernel<<<waves * 32 / 256, 256, 0, stream>>>(qbuf, bm, sel);
    }
    // 5) flash attention with MoBA mask (WMMA)
    {
        int waves = NH * (S / 16);              // 4096 waves
        flash_kernel<<<waves * 32 / 128, 128, 0, stream>>>(qbuf, kbuf, vT, sel, attn);
    }
    // 6) output projection -> fp32 d_out (WMMA)
    {
        int waves = (S / 32) * (DIM / 64);      // 4096
        gemm_bf16_kernel<true><<<waves * 32 / 128, 128, 0, stream>>>(attn, woT, out, S, DIM, DIM);
    }
    (void)in_sizes; (void)n_in; (void)out_size; (void)ws_size;
}
